// CrossModal_7206955123204
// MI455X (gfx1250) — matrix operand order repaired and checked by reading, not compiled, not run on hardware
//
#include <hip/hip_runtime.h>
#include <stdint.h>

#define Bc   32
#define Lc   512
#define Dc   768
#define MAXN 128
#define NLAY 3

typedef __attribute__((ext_vector_type(16))) _Float16 v16h;
typedef __attribute__((ext_vector_type(8)))  float    v8f;

union Frag { v16h h; uint32_t u[8]; };

// A-matrix 16x32 f16 layout (ISA 7.12.2): lane m = lane&15, half = lane>>4.
// VGPR v<4: K = 2v,2v+1 (+8 for upper half); v>=4: K = 16+2(v-4) (+8 upper).
__device__ __forceinline__ int a_koff(int v, int hi) {
  return (v < 4) ? (2 * v + 8 * hi) : (16 + 2 * (v - 4) + 8 * hi);
}
// B-matrix 32x16: lane n = lane&15; same K order as A:
// VGPR v<4: K = 2v,2v+1 (+8 for upper half); v>=4: K = 16+2(v-4) (+8 upper).
__device__ __forceinline__ int b_koff(int v, int hi) { return (v < 4) ? (2 * v + 8 * hi) : (16 + 2 * (v - 4) + 8 * hi); }

#define WMMA_F16(A, B, C) \
  __builtin_amdgcn_wmma_f32_16x16x32_f16(false, (A).h, false, (B).h, (short)0, (C), false, false)

__device__ __forceinline__ float wave_max(float v) {
#pragma unroll
  for (int o = 16; o >= 1; o >>= 1) v = fmaxf(v, __shfl_xor(v, o, 32));
  return v;
}
__device__ __forceinline__ float wave_sum(float v) {
#pragma unroll
  for (int o = 16; o >= 1; o >>= 1) v += __shfl_xor(v, o, 32);
  return v;
}

// ---------------- init kernels ----------------
__global__ __launch_bounds__(256) void k_f32_to_f16(const float* __restrict__ src,
                                                    _Float16* __restrict__ dst, int n) {
  int i = blockIdx.x * 256 + threadIdx.x;
  if (i < n) dst[i] = (_Float16)src[i];
}

// dst[l][n][k] = src[l][k][n]  (transpose-convert weights so B-fragment loads
// become contiguous dword loads)
__global__ __launch_bounds__(256) void k_wT_to_f16(const float* __restrict__ src,
                                                   _Float16* __restrict__ dst, int n) {
  int i = blockIdx.x * 256 + threadIdx.x;
  if (i >= n) return;
  int k = i % Dc;
  int nn = (i / Dc) % Dc;
  int l = i / (Dc * Dc);
  dst[i] = (_Float16)src[(size_t)l * Dc * Dc + (size_t)k * Dc + nn];
}

__global__ void k_nper(const unsigned char* __restrict__ mask, float* __restrict__ nper) {
  int b = threadIdx.x;
  if (b >= Bc) return;
  float s = 0.f;
  for (int m = 0; m < MAXN; ++m) s += mask[b * MAXN + m] ? 1.f : 0.f;
  nper[b] = s;
}

__global__ __launch_bounds__(256) void k_edge_cnt(const int* __restrict__ dst,
                                                  float* __restrict__ cnt, int E) {
  int i = blockIdx.x * 256 + threadIdx.x;
  if (i < E) atomicAdd(&cnt[dst[i]], 1.f);
}

__global__ __launch_bounds__(256) void k_invcnt(float* __restrict__ cnt, int n) {
  int i = blockIdx.x * 256 + threadIdx.x;
  if (i < n) cnt[i] = 1.f / fmaxf(cnt[i], 1.f);
}

// ---------------- per-layer kernels ----------------
__global__ __launch_bounds__(256) void k_edge_scatter(const float* __restrict__ x,
                                                      const int* __restrict__ src,
                                                      const int* __restrict__ dst,
                                                      float* __restrict__ msg, int total) {
  int i = blockIdx.x * 256 + threadIdx.x;
  if (i >= total) return;
  int e = i / Dc, d = i % Dc;
  atomicAdd(&msg[(size_t)dst[e] * Dc + d], x[(size_t)src[e] * Dc + d]);
}

__global__ __launch_bounds__(256) void k_prep_halves(const float* __restrict__ msg,
                                                     const float* __restrict__ inv,
                                                     const float* __restrict__ x,
                                                     _Float16* __restrict__ msgh,
                                                     _Float16* __restrict__ xh, int total) {
  int i = blockIdx.x * 256 + threadIdx.x;
  if (i >= total) return;
  int n = i / Dc;
  msgh[i] = (_Float16)(msg[i] * inv[n]);
  xh[i]   = (_Float16)x[i];
}

// y = relu(x @ Ws + msg @ Wn + b). 2x2 register-blocked WMMA: each wave owns a
// 32x32 output block (4 tiles); per K-step: 8 fragment loads -> 8 WMMAs.
__global__ __launch_bounds__(256) void k_gemm_gnn(const _Float16* __restrict__ xh,
                                                  const _Float16* __restrict__ mh,
                                                  const _Float16* __restrict__ wsT,
                                                  const _Float16* __restrict__ wnT,
                                                  const float* __restrict__ bias,
                                                  float* __restrict__ y, int Nrows) {
  const int wave = blockIdx.x * 8 + (threadIdx.x >> 5);
  const int lane = threadIdx.x & 31;
  const int macroN = Dc / 32;  // 24
  const int mM = wave / macroN;
  const int mN = wave % macroN;
  if (mM * 32 >= Nrows) return;
  const int hi = lane >> 4;
  const int l15 = lane & 15;
  const int r0 = mM * 32 + l15;
  const int r1 = (r0 + 16 < Nrows) ? (r0 + 16) : (Nrows - 1);
  const int c0 = mN * 32 + l15;
  const int c1 = c0 + 16;
  const _Float16* ax0 = xh + (size_t)r0 * Dc;
  const _Float16* ax1 = xh + (size_t)r1 * Dc;
  const _Float16* am0 = mh + (size_t)r0 * Dc;
  const _Float16* am1 = mh + (size_t)r1 * Dc;
  const _Float16* bs0 = wsT + (size_t)c0 * Dc;
  const _Float16* bs1 = wsT + (size_t)c1 * Dc;
  const _Float16* bn0 = wnT + (size_t)c0 * Dc;
  const _Float16* bn1 = wnT + (size_t)c1 * Dc;
  v8f a00 = {}, a01 = {}, a10 = {}, a11 = {};
  for (int k0 = 0; k0 < Dc; k0 += 32) {
    Frag fx0, fx1, fg0, fg1, fs0, fs1, fn0, fn1;
#pragma unroll
    for (int v = 0; v < 8; ++v) {
      int ka = k0 + a_koff(v, hi);
      int kb = k0 + b_koff(v, hi);
      fx0.u[v] = *(const uint32_t*)(ax0 + ka);
      fx1.u[v] = *(const uint32_t*)(ax1 + ka);
      fg0.u[v] = *(const uint32_t*)(am0 + ka);
      fg1.u[v] = *(const uint32_t*)(am1 + ka);
      fs0.u[v] = *(const uint32_t*)(bs0 + kb);
      fs1.u[v] = *(const uint32_t*)(bs1 + kb);
      fn0.u[v] = *(const uint32_t*)(bn0 + kb);
      fn1.u[v] = *(const uint32_t*)(bn1 + kb);
    }
    a00 = WMMA_F16(fx0, fs0, a00);
    a00 = WMMA_F16(fg0, fn0, a00);
    a01 = WMMA_F16(fx0, fs1, a01);
    a01 = WMMA_F16(fg0, fn1, a01);
    a10 = WMMA_F16(fx1, fs0, a10);
    a10 = WMMA_F16(fg1, fn0, a10);
    a11 = WMMA_F16(fx1, fs1, a11);
    a11 = WMMA_F16(fg1, fn1, a11);
  }
  // Tile-level bounds: row < rbase+16 always, so the whole tile is interior
  // iff rbase+16 <= Nrows -> unpredicated store fast path (no per-element exec
  // juggling; always taken when Nrows % 32 == 0).
  auto store_tile = [&](const v8f& acc, int rbase, int cbase) {
    const int col = cbase + l15;
    if (rbase + 16 <= Nrows) {
#pragma unroll
      for (int v = 0; v < 8; ++v) {
        int row = rbase + v + 8 * hi;  // C/D layout: VGPR v -> M = v + 8*half
        float val = acc[v] + bias[col];
        y[(size_t)row * Dc + col] = val > 0.f ? val : 0.f;
      }
    } else {
#pragma unroll
      for (int v = 0; v < 8; ++v) {
        int row = rbase + v + 8 * hi;
        if (row < Nrows) {
          float val = acc[v] + bias[col];
          y[(size_t)row * Dc + col] = val > 0.f ? val : 0.f;
        }
      }
    }
  };
  store_tile(a00, mM * 32, mN * 32);
  store_tile(a01, mM * 32, mN * 32 + 16);
  store_tile(a10, mM * 32 + 16, mN * 32);
  store_tile(a11, mM * 32 + 16, mN * 32 + 16);
}

__global__ __launch_bounds__(256) void k_gather_xp(const float* __restrict__ x,
                                                   const int* __restrict__ pidx,
                                                   const unsigned char* __restrict__ mask,
                                                   float* __restrict__ xpf,
                                                   _Float16* __restrict__ xph) {
  int i = blockIdx.x * 256 + threadIdx.x;
  if (i >= Bc * MAXN * Dc) return;
  int d = i % Dc;
  int bm = i / Dc;
  float val = mask[bm] ? x[(size_t)pidx[bm] * Dc + d] : 0.f;
  xpf[i] = val;
  xph[i] = (_Float16)val;
}

// S[b,m,l] = sum_d xp[b,m,d] * text[b,l,d], 2x2 register-blocked.
__global__ __launch_bounds__(256) void k_gemm_scores(const _Float16* __restrict__ xph,
                                                     const _Float16* __restrict__ th,
                                                     float* __restrict__ S) {
  const int wave = blockIdx.x * 8 + (threadIdx.x >> 5);
  const int lane = threadIdx.x & 31;
  const int macroM = MAXN / 32;  // 4
  const int macroL = Lc / 32;    // 16
  const int perB = macroM * macroL;
  const int b = wave / perB;
  if (b >= Bc) return;
  const int t = wave % perB;
  const int mM = t / macroL;
  const int mL = t % macroL;
  const int hi = lane >> 4;
  const int l15 = lane & 15;
  const _Float16* A0 = xph + ((size_t)b * MAXN + mM * 32 + l15) * Dc;
  const _Float16* A1 = A0 + (size_t)16 * Dc;
  const _Float16* B0 = th + ((size_t)b * Lc + mL * 32 + l15) * Dc;
  const _Float16* B1 = B0 + (size_t)16 * Dc;
  v8f a00 = {}, a01 = {}, a10 = {}, a11 = {};
  for (int k0 = 0; k0 < Dc; k0 += 32) {
    Frag fa0, fa1, fb0, fb1;
#pragma unroll
    for (int v = 0; v < 8; ++v) {
      int ka = k0 + a_koff(v, hi);
      int kb = k0 + b_koff(v, hi);
      fa0.u[v] = *(const uint32_t*)(A0 + ka);
      fa1.u[v] = *(const uint32_t*)(A1 + ka);
      fb0.u[v] = *(const uint32_t*)(B0 + kb);
      fb1.u[v] = *(const uint32_t*)(B1 + kb);
    }
    a00 = WMMA_F16(fa0, fb0, a00);
    a01 = WMMA_F16(fa0, fb1, a01);
    a10 = WMMA_F16(fa1, fb0, a10);
    a11 = WMMA_F16(fa1, fb1, a11);
  }
  auto store_tile = [&](const v8f& acc, int rbase, int cbase) {
#pragma unroll
    for (int v = 0; v < 8; ++v) {
      int row = rbase + v + 8 * hi;
      int col = cbase + l15;
      S[((size_t)b * MAXN + row) * Lc + col] = acc[v];
    }
  };
  store_tile(a00, mM * 32, mL * 32);
  store_tile(a01, mM * 32, mL * 32 + 16);
  store_tile(a10, mM * 32 + 16, mL * 32);
  store_tile(a11, mM * 32 + 16, mL * 32 + 16);
}

// g2t: softmax over l per row m; accumulate w[b,l] += mask_m * a[m,l]
__global__ __launch_bounds__(256) void k_row_softmax(const float* __restrict__ S,
                                                     const unsigned char* __restrict__ mask,
                                                     float* __restrict__ w) {
  int wave = blockIdx.x * 8 + (threadIdx.x >> 5);
  int lane = threadIdx.x & 31;
  if (wave >= Bc * MAXN) return;
  int b = wave / MAXN;
  int m = wave % MAXN;
  const float* row = S + (size_t)wave * Lc;
  float vals[16];
  float mx = -3.4e38f;
#pragma unroll
  for (int j = 0; j < 16; ++j) { vals[j] = row[lane + 32 * j]; mx = fmaxf(mx, vals[j]); }
  mx = wave_max(mx);
  float s = 0.f;
#pragma unroll
  for (int j = 0; j < 16; ++j) { vals[j] = __expf(vals[j] - mx); s += vals[j]; }
  s = wave_sum(s);
  if (!mask[b * MAXN + m]) return;
  float invs = 1.f / s;
#pragma unroll
  for (int j = 0; j < 16; ++j) atomicAdd(&w[(size_t)b * Lc + lane + 32 * j], vals[j] * invs);
}

// t2g: masked softmax over m per column l; accumulate v[b,m] += p[m,l]
__global__ __launch_bounds__(256) void k_col_softmax(const float* __restrict__ S,
                                                     const unsigned char* __restrict__ mask,
                                                     float* __restrict__ vv) {
  int wave = blockIdx.x * 8 + (threadIdx.x >> 5);
  int lane = threadIdx.x & 31;
  if (wave >= Bc * Lc) return;
  int b = wave / Lc;
  int l = wave % Lc;
  float vals[4];
  int mk[4];
  float mx = -3.4e38f;
#pragma unroll
  for (int j = 0; j < 4; ++j) {
    int m = lane + 32 * j;
    mk[j] = mask[b * MAXN + m] ? 1 : 0;
    vals[j] = mk[j] ? S[((size_t)b * MAXN + m) * Lc + l] : -1e9f;
    mx = fmaxf(mx, vals[j]);
  }
  mx = wave_max(mx);
  float s = 0.f;
#pragma unroll
  for (int j = 0; j < 4; ++j) { vals[j] = __expf(vals[j] - mx); s += vals[j]; }
  s = wave_sum(s);
  float invs = 1.f / s;
#pragma unroll
  for (int j = 0; j < 4; ++j)
    if (mk[j]) atomicAdd(&vv[b * MAXN + lane + 32 * j], vals[j] * invs);
}

// pooled[b,d] = (sum_l w[b,l]*text[b,l,d] + sum_m v[b,m]*xp[b,m,d]) / (n_per+L)
__global__ __launch_bounds__(256) void k_pooled(const float* __restrict__ w,
                                                const float* __restrict__ vv,
                                                const float* __restrict__ text,
                                                const float* __restrict__ xpf,
                                                const float* __restrict__ nper,
                                                float* __restrict__ cms) {
  int i = blockIdx.x * 256 + threadIdx.x;
  if (i >= Bc * Dc) return;
  int b = i / Dc, d = i % Dc;
  float p = 0.f;
  const float* tb = text + (size_t)b * Lc * Dc + d;
  for (int l = 0; l < Lc; ++l) p += w[(size_t)b * Lc + l] * tb[(size_t)l * Dc];
  const float* xb = xpf + (size_t)b * MAXN * Dc + d;
  for (int m = 0; m < MAXN; ++m) p += vv[b * MAXN + m] * xb[(size_t)m * Dc];
  cms[i] += p / (nper[b] + (float)Lc);
}

// ---------------- final kernels ----------------
__global__ __launch_bounds__(256) void k_gsum(const float* __restrict__ x,
                                              const int* __restrict__ batch,
                                              float* __restrict__ gsum, int total) {
  int i = blockIdx.x * 256 + threadIdx.x;
  if (i >= total) return;
  int n = i / Dc, d = i % Dc;
  atomicAdd(&gsum[(size_t)batch[n] * Dc + d], x[i]);
}

__global__ __launch_bounds__(256) void k_out(const float* __restrict__ text,
                                             const float* __restrict__ gsum,
                                             const float* __restrict__ nper,
                                             const float* __restrict__ cms,
                                             const float* __restrict__ aw,
                                             const float* __restrict__ ab,
                                             const float* __restrict__ z,
                                             const float* __restrict__ dmask,
                                             const float* __restrict__ gmask,
                                             float* __restrict__ out) {
  int i = blockIdx.x * 256 + threadIdx.x;
  if (i >= Bc * Dc) return;
  int b = i / Dc, d = i % Dc;
  float invn = 1.f / fmaxf(nper[b], 1.f);
  float acc = ab[d];
  for (int k = 0; k < Dc; ++k)
    acc += (gsum[(size_t)b * Dc + k] * invn) * aw[(size_t)k * Dc + d];
  float og = tanhf(acc);
  float ot = text[(size_t)b * Lc * Dc + d];
  float cm = cms[i] * (1.f / (float)NLAY);
  float dm = dmask[b], gm = gmask[b], dg = dm * gm, zz = z[d];
  out[(size_t)b * (3 * Dc) + d]          = ot * dm + (1.f - dm) * zz;
  out[(size_t)b * (3 * Dc) + Dc + d]     = og * gm + (1.f - gm) * zz;
  out[(size_t)b * (3 * Dc) + 2 * Dc + d] = cm * dg + (1.f - dg) * zz;
}

extern "C" void kernel_launch(void* const* d_in, const int* in_sizes, int n_in,
                              void* d_out, int out_size, void* d_ws, size_t ws_size,
                              hipStream_t stream) {
  const float* x_in  = (const float*)d_in[0];
  const float* text  = (const float*)d_in[1];
  const float* dmask = (const float*)d_in[2];
  const float* gmask = (const float*)d_in[3];
  const float* gws   = (const float*)d_in[4];
  const float* gwn   = (const float*)d_in[5];
  const float* gb    = (const float*)d_in[6];
  const float* aw    = (const float*)d_in[7];
  const float* ab    = (const float*)d_in[8];
  const float* zemb  = (const float*)d_in[9];
  const int* batch   = (const int*)d_in[10];
  const int* pidx    = (const int*)d_in[11];
  const unsigned char* pmask = (const unsigned char*)d_in[12];
  const int* esrc    = (const int*)d_in[13];
  const int* edst    = (const int*)d_in[14];
  float* out = (float*)d_out;

  const int N = in_sizes[0] / Dc;
  const int E = in_sizes[13];

  char* p = (char*)d_ws;
  auto carve = [&](size_t bytes) -> void* {
    void* r = (void*)p;
    p += (bytes + 255) & ~(size_t)255;
    return r;
  };
  float*    xA    = (float*)carve((size_t)N * Dc * 4);
  float*    xB    = (float*)carve((size_t)N * Dc * 4);
  float*    msg   = (float*)carve((size_t)N * Dc * 4);
  _Float16* xh    = (_Float16*)carve((size_t)N * Dc * 2);
  _Float16* msgh  = (_Float16*)carve((size_t)N * Dc * 2);
  _Float16* texth = (_Float16*)carve((size_t)Bc * Lc * Dc * 2);
  _Float16* wsT   = (_Float16*)carve((size_t)NLAY * Dc * Dc * 2);
  _Float16* wnT   = (_Float16*)carve((size_t)NLAY * Dc * Dc * 2);
  float*    xpf   = (float*)carve((size_t)Bc * MAXN * Dc * 4);
  _Float16* xph   = (_Float16*)carve((size_t)Bc * MAXN * Dc * 2);
  float*    S     = (float*)carve((size_t)Bc * MAXN * Lc * 4);
  float*    w     = (float*)carve((size_t)Bc * Lc * 4);
  float*    vv    = (float*)carve((size_t)Bc * MAXN * 4);
  float*    inv   = (float*)carve((size_t)N * 4);
  float*    nper  = (float*)carve((size_t)Bc * 4);
  float*    cms   = (float*)carve((size_t)Bc * Dc * 4);
  float*    gsum  = (float*)carve((size_t)Bc * Dc * 4);

  auto blocks = [](long long n) { return (int)((n + 255) / 256); };

  // ---- init ----
  hipMemcpyAsync(xA, x_in, (size_t)N * Dc * 4, hipMemcpyDeviceToDevice, stream);
  hipMemsetAsync(inv, 0, (size_t)N * 4, stream);
  hipMemsetAsync(cms, 0, (size_t)Bc * Dc * 4, stream);
  k_f32_to_f16<<<blocks((long long)Bc * Lc * Dc), 256, 0, stream>>>(text, texth, Bc * Lc * Dc);
  k_wT_to_f16<<<blocks((long long)NLAY * Dc * Dc), 256, 0, stream>>>(gws, wsT, NLAY * Dc * Dc);
  k_wT_to_f16<<<blocks((long long)NLAY * Dc * Dc), 256, 0, stream>>>(gwn, wnT, NLAY * Dc * Dc);
  k_nper<<<1, 32, 0, stream>>>(pmask, nper);
  k_edge_cnt<<<blocks(E), 256, 0, stream>>>(edst, inv, E);
  k_invcnt<<<blocks(N), 256, 0, stream>>>(inv, N);

  const int macroM = (N + 31) / 32;
  const int gemmBlocks = (macroM * (Dc / 32) + 7) / 8;
  const int sBlocks = (Bc * (MAXN / 32) * (Lc / 32) + 7) / 8;

  float* xcur = xA;
  float* xnext = xB;
  for (int layer = 0; layer < NLAY; ++layer) {
    hipMemsetAsync(msg, 0, (size_t)N * Dc * 4, stream);
    k_edge_scatter<<<blocks((long long)E * Dc), 256, 0, stream>>>(xcur, esrc, edst, msg, E * Dc);
    k_prep_halves<<<blocks((long long)N * Dc), 256, 0, stream>>>(msg, inv, xcur, msgh, xh, N * Dc);
    k_gemm_gnn<<<gemmBlocks, 256, 0, stream>>>(xh, msgh, wsT + (size_t)layer * Dc * Dc,
                                               wnT + (size_t)layer * Dc * Dc,
                                               gb + (size_t)layer * Dc, xnext, N);
    k_gather_xp<<<blocks((long long)Bc * MAXN * Dc), 256, 0, stream>>>(xnext, pidx, pmask, xpf, xph);
    k_gemm_scores<<<sBlocks, 256, 0, stream>>>(xph, texth, S);
    hipMemsetAsync(w, 0, (size_t)Bc * Lc * 4, stream);
    hipMemsetAsync(vv, 0, (size_t)Bc * MAXN * 4, stream);
    k_row_softmax<<<(Bc * MAXN + 7) / 8, 256, 0, stream>>>(S, pmask, w);
    k_col_softmax<<<(Bc * Lc + 7) / 8, 256, 0, stream>>>(S, pmask, vv);
    k_pooled<<<blocks(Bc * Dc), 256, 0, stream>>>(w, vv, text, xpf, nper, cms);
    float* tmp = xcur; xcur = xnext; xnext = tmp;
  }

  hipMemsetAsync(gsum, 0, (size_t)Bc * Dc * 4, stream);
  k_gsum<<<blocks((long long)N * Dc), 256, 0, stream>>>(xcur, batch, gsum, N * Dc);
  k_out<<<blocks(Bc * Dc), 256, 0, stream>>>(text, gsum, nper, cms, aw, ab, zemb, dmask, gmask, out);
}
